// FeedForwardQuantum_65481071399837
// MI455X (gfx1250) — matrix operand order repaired and checked by reading, not compiled, not run on hardware
//
#include <hip/hip_runtime.h>

typedef _Float16 v16h __attribute__((ext_vector_type(16)));
typedef _Float16 v8h  __attribute__((ext_vector_type(8)));
typedef _Float16 h2   __attribute__((ext_vector_type(2)));
typedef float    v8f  __attribute__((ext_vector_type(8)));
typedef unsigned int u32x4 __attribute__((ext_vector_type(4)));
typedef int      i32x4 __attribute__((ext_vector_type(4)));
typedef int      i32x8 __attribute__((ext_vector_type(8)));

namespace {
constexpr int E_DIM  = 1024;
constexpr int F_DIM  = 4096;
constexpr int M_TILE = 64;        // rows per block; 8 waves each span all of M
constexpr int KC     = 32;        // K (=F) chunk per TDM stage
constexpr int NCH    = F_DIM / KC;          // 128 chunks
constexpr int KSTR   = 40;        // LDS row stride in halves (32 data + 8 pad)
// LDS layout (single symbol => offset 0 of workgroup LDS)
constexpr unsigned W2SZ  = E_DIM * KSTR * 2;      // 81920 B per w2 buffer
constexpr unsigned HSZ   = M_TILE * KSTR * 2;     //  5120 B per h buffer
constexpr unsigned W2OFF0 = 0;
constexpr unsigned W2OFF1 = W2SZ;
constexpr unsigned HOFF0  = 2 * W2SZ;
constexpr unsigned HOFF1  = 2 * W2SZ + HSZ;
constexpr unsigned QOFF   = 2 * W2SZ + 2 * HSZ;
constexpr unsigned SMEM_BYTES = QOFF + M_TILE * 8 * 4;   // 176128 B
constexpr size_t   W1C_WS_OFF = (size_t)F_DIM * E_DIM * 2;  // 8 MB: w2h region
}

// ---- prologue: w2 -> f16 in ws; w1c = w1 * cos(theta) in ws ----
__global__ __launch_bounds__(256) void ffq_prep_kernel(
    const float* __restrict__ w2, const float* __restrict__ w1,
    const float* __restrict__ theta, _Float16* __restrict__ w2h,
    float* __restrict__ w1c)
{
  const int b = blockIdx.x, t = threadIdx.x;
  if (b < 4096) {                    // 4096*1024 = 4194304 = E*F elements
    const size_t i = (size_t)b * 1024 + (size_t)t * 4;
    const float4 v = *(const float4*)(w2 + i);
    *(h2*)(w2h + i)     = h2{(_Float16)v.x, (_Float16)v.y};
    *(h2*)(w2h + i + 2) = h2{(_Float16)v.z, (_Float16)v.w};
  } else {                           // 32*1024 = 32768 = F*8 elements
    const int j = (b - 4096) * 1024 + t * 4;
    const float4 v = *(const float4*)(w1 + j);
    const int i0 = j & 7;            // 0 or 4
    w1c[j + 0] = v.x * __builtin_cosf(theta[i0 + 0]);
    w1c[j + 1] = v.y * __builtin_cosf(theta[i0 + 1]);
    w1c[j + 2] = v.z * __builtin_cosf(theta[i0 + 2]);
    w1c[j + 3] = v.w * __builtin_cosf(theta[i0 + 3]);
  }
}

__global__ __launch_bounds__(256) void ffq_wmma_f16_kernel(
    const float* __restrict__ x,
    const float* __restrict__ b1, const float* __restrict__ b2,
    const _Float16* __restrict__ w2h, const float* __restrict__ w1c,
    float* __restrict__ out)
{
  __shared__ __align__(128) unsigned char smem[SMEM_BYTES];
  float* const qs = (float*)(smem + QOFF);

  const int tid  = threadIdx.x;
  const int lane = tid & 31;
  const int lm   = lane & 15;
  const int half = lane >> 4;
  const int wave = tid >> 5;
  const int mBase = blockIdx.x * M_TILE;
  const int nW    = wave * 128;                 // wave's N slice
  // scalar (wave-uniform) wave-0 predicate so TDM is issued by exactly one wave
  const bool isWave0 = (__builtin_amdgcn_readfirstlane(tid) >> 5) == 0;

  // ---- q = cos(x[:, :8]) (cos(theta) folded into w1c) ----
  if (tid < 128) {
    const int m = tid >> 1, i4 = (tid & 1) * 4;
    const float4 xv = *(const float4*)(x + (size_t)(mBase + m) * E_DIM + i4);
    float4 qv;
    qv.x = __builtin_cosf(xv.x);  qv.y = __builtin_cosf(xv.y);
    qv.z = __builtin_cosf(xv.z);  qv.w = __builtin_cosf(xv.w);
    *(float4*)(qs + m * 8 + i4) = qv;
  }

  // ---- TDM issue: stage w2h[0:1024, kBase:kBase+32] -> LDS (stride-padded) ----
  auto tdm_issue = [&](int kBase, unsigned ldsOff) {
    const unsigned long long ga =
        (unsigned long long)(const void*)w2h + (unsigned long long)kBase * 2ull;
    u32x4 g0;
    g0[0] = 1u;                                   // count=1, user mode
    g0[1] = ldsOff;                               // lds_addr
    g0[2] = (unsigned)(ga & 0xFFFFFFFFull);       // global_addr lo
    g0[3] = (unsigned)((ga >> 32) & 0x1FFFFFFull) | 0x80000000u;  // addr hi | type=2
    i32x8 g1;
    g1[0] = (1 << 16) | (1 << 20) | (3 << 22) | (3 << 25);  // 2B, pad_en, 16dw interval, 4dw pad
    g1[1] = (F_DIM & 0xFFFF) << 16;               // tensor_dim0 = 4096 (lo16)
    g1[2] = (E_DIM & 0xFFFF) << 16;               // tensor_dim1 = 1024 (lo16)
    g1[3] = KC << 16;                             // tile_dim0 = 32
    g1[4] = E_DIM;                                // tile_dim1 = 1024, tile_dim2 = 0
    g1[5] = F_DIM;                                // tensor_dim0_stride = 4096
    g1[6] = 0; g1[7] = 0;
    i32x4 g2; g2[0] = 1; g2[1] = 0; g2[2] = 0; g2[3] = 0;   // tensor_dim2=1 (degenerate)
    i32x4 g3; g3[0] = 0; g3[1] = 0; g3[2] = 0; g3[3] = 0;
#if defined(__clang_major__) && __clang_major__ >= 23
    i32x8 gz = {};
    __builtin_amdgcn_tensor_load_to_lds(g0, g1, g2, g3, gz, 0);
#else
    __builtin_amdgcn_tensor_load_to_lds(g0, g1, g2, g3, 0);
#endif
  };

  // ---- GEMM1 (VALU, tiny): h[m, kBase+k] = relu(cos(x).w1c + b1) -> f16 LDS ----
  const int hk  = (tid & 15) * 2;     // k pair within chunk
  const int hmb = tid >> 4;           // row base, step 16
  auto compute_h = [&](int kBase, _Float16* hbuf) {
    const float* w1r = w1c + (size_t)(kBase + hk) * 8;
    const float4 a0 = *(const float4*)(w1r);
    const float4 a1 = *(const float4*)(w1r + 4);
    const float4 c0 = *(const float4*)(w1r + 8);
    const float4 c1 = *(const float4*)(w1r + 12);
    const float bb0 = b1[kBase + hk], bb1 = b1[kBase + hk + 1];
    #pragma unroll
    for (int j = 0; j < 4; ++j) {
      const int m = hmb + 16 * j;
      const float* qr = qs + m * 8;
      float s0 = bb0 + qr[0]*a0.x + qr[1]*a0.y + qr[2]*a0.z + qr[3]*a0.w
                     + qr[4]*a1.x + qr[5]*a1.y + qr[6]*a1.z + qr[7]*a1.w;
      float s1 = bb1 + qr[0]*c0.x + qr[1]*c0.y + qr[2]*c0.z + qr[3]*c0.w
                     + qr[4]*c1.x + qr[5]*c1.y + qr[6]*c1.z + qr[7]*c1.w;
      *(h2*)(hbuf + m * KSTR + hk) =
          h2{(_Float16)fmaxf(s0, 0.0f), (_Float16)fmaxf(s1, 0.0f)};
    }
  };

  __syncthreads();                       // qs visible
  // prologue: fill buffer 0
  if (isWave0) tdm_issue(0, W2OFF0);
  compute_h(0, (_Float16*)(smem + HOFF0));
  if (isWave0) __builtin_amdgcn_s_wait_tensorcnt(0);
  __syncthreads();                       // buffer 0 ready

  v8f acc[4][8] = {};                    // 4 M-tiles x 8 N-tiles per wave

  for (int c = 0; c < NCH; ++c) {
    const int p = c & 1;
    const _Float16* const w2p = (const _Float16*)(smem + (p ? W2OFF1 : W2OFF0));
    const _Float16* const hp  = (const _Float16*)(smem + (p ? HOFF1  : HOFF0));
    _Float16* const hn        = (_Float16*)(smem + (p ? HOFF0 : HOFF1));

    if (c + 1 < NCH && isWave0) tdm_issue((c + 1) * KC, p ? W2OFF0 : W2OFF1);

    // A fragments: 16-bit A 16x32 layout (two 16B runs at k=8*half, 16+8*half)
    v16h af[4];
    #pragma unroll
    for (int mt = 0; mt < 4; ++mt) {
      const _Float16* ar = hp + (mt * 16 + lm) * KSTR + 8 * half;
      const v8h lo = *(const v8h*)(ar);
      const v8h hi = *(const v8h*)(ar + 16);
      af[mt] = __builtin_shufflevector(lo, hi, 0,1,2,3,4,5,6,7,8,9,10,11,12,13,14,15);
    }
    #pragma unroll
    for (int nt = 0; nt < 8; ++nt) {
      // B fragment: 32x16 layout, two 16B runs at k = 8*half, 16+8*half
      const _Float16* br = w2p + (nW + nt * 16 + lm) * KSTR + 8 * half;
      const v8h blo = *(const v8h*)(br);
      const v8h bhi = *(const v8h*)(br + 16);
      const v16h bf = __builtin_shufflevector(blo, bhi, 0,1,2,3,4,5,6,7,8,9,10,11,12,13,14,15);
      #pragma unroll
      for (int mt = 0; mt < 4; ++mt)
        acc[mt][nt] = __builtin_amdgcn_wmma_f32_16x16x32_f16(
            false, af[mt], false, bf, (short)0, acc[mt][nt], false, false);
    }

    if (c + 1 < NCH) compute_h((c + 1) * KC, hn);
    if (isWave0) __builtin_amdgcn_s_wait_tensorcnt(0);
    __syncthreads();
  }

  // ---- epilogue: out = acc + b2 (C layout: N=lm, M=r+8*half) ----
  #pragma unroll
  for (int nt = 0; nt < 8; ++nt) {
    const int n = nW + nt * 16 + lm;
    const float bias = b2[n];
    #pragma unroll
    for (int mt = 0; mt < 4; ++mt) {
      #pragma unroll
      for (int r = 0; r < 8; ++r) {
        const int m = mBase + mt * 16 + r + 8 * half;
        out[(size_t)m * E_DIM + n] = acc[mt][nt][r] + bias;
      }
    }
  }
}

extern "C" void kernel_launch(void* const* d_in, const int* in_sizes, int n_in,
                              void* d_out, int out_size, void* d_ws, size_t ws_size,
                              hipStream_t stream) {
  const float* x     = (const float*)d_in[0];
  const float* theta = (const float*)d_in[1];
  const float* w1    = (const float*)d_in[2];
  const float* b1    = (const float*)d_in[3];
  const float* w2    = (const float*)d_in[4];
  const float* b2    = (const float*)d_in[5];
  float* out = (float*)d_out;

  _Float16* w2h = (_Float16*)d_ws;                       // 8 MB f16 w2
  float*    w1c = (float*)((char*)d_ws + W1C_WS_OFF);    // 128 KB w1*cos(theta)

  ffq_prep_kernel<<<4096 + 32, 256, 0, stream>>>(w2, w1, theta, w2h, w1c);

  const int Mtotal = in_sizes[0] / E_DIM;                // 32768
  ffq_wmma_f16_kernel<<<Mtotal / M_TILE, 256, 0, stream>>>(x, b1, b2, w2h, w1c, out);
}
